// WeightedGraphLayer_35424890257853
// MI455X (gfx1250) — matrix operand order repaired and checked by reading, not compiled, not run on hardware
//
#include <hip/hip_runtime.h>
#include <hip/hip_bf16.h>

typedef __attribute__((ext_vector_type(16))) _Float16 v16h;
typedef __attribute__((ext_vector_type(8)))  _Float16 v8h;
typedef __attribute__((ext_vector_type(8)))  float    v8f;

#define B_   16
#define N_   256
#define K_   64
#define D_   128
#define OUT_ 128
#define CD_  5

#define K1P  160   // 135 padded to 160 (5 k-steps of 32)
#define K2P  128   // 4 k-steps
#define KNP  288   // 261 padded to 288 (9 k-steps)

// ---------- async global->LDS staging (CDNA5 ASYNCcnt path), with fallback ----------
#if defined(__AMDGCN__) && __has_builtin(__builtin_amdgcn_global_load_async_to_lds_b128) && \
    __has_builtin(__builtin_amdgcn_s_wait_asynccnt)
#define HAVE_ASYNC_LDS 1
#else
#define HAVE_ASYNC_LDS 0
#endif

#if HAVE_ASYNC_LDS
// builtin expects: (int4 AS1* global_src, int4 AS3* lds_dst, imm offset, imm cpol)
typedef int v4i_t __attribute__((vector_size(16)));
typedef __attribute__((address_space(1))) v4i_t g1_v4i;
typedef __attribute__((address_space(3))) v4i_t l3_v4i;

__device__ inline g1_v4i* as_global(const void* p) {
  return (g1_v4i*)(unsigned long long)(uintptr_t)p;
}
// flat LDS pointers carry the LDS byte offset in their low 32 bits (ISA 10.2)
__device__ inline l3_v4i* as_lds(void* p) {
  return (l3_v4i*)(unsigned)(uintptr_t)p;
}
#endif

// Copy `bytes` (multiple of 16) from global to LDS with all 256 threads.
__device__ inline void stage_g2l(const void* gsrc, void* ldst, int bytes, int tid) {
#if HAVE_ASYNC_LDS
  for (int off = tid * 16; off < bytes; off += 256 * 16) {
    __builtin_amdgcn_global_load_async_to_lds_b128(
        as_global((const char*)gsrc + off), as_lds((char*)ldst + off), 0, 0);
  }
#else
  const uint4* src = (const uint4*)gsrc;
  uint4* dst = (uint4*)ldst;
  for (int i = tid; i < bytes / 16; i += 256) dst[i] = src[i];
#endif
}

__device__ inline void stage_wait() {
#if HAVE_ASYNC_LDS
  __builtin_amdgcn_s_wait_asynccnt(0);
#endif
}

// ---------- WMMA fragment loaders (ISA 7.12.2 layouts, wave32) ----------
// A (16x32 f16, row-major source): lane L holds row M=L%16;
//   halves 0..7  = K (col0 + 8*(L/16)) + 0..7
//   halves 8..15 = K (col0 + 8*(L/16)) + 16..23
// Two aligned 16-byte LDS loads -> ds_load_b128 x2.
__device__ inline v16h frag_A(const _Float16* s, int stride, int row0, int col0, int lane) {
  const _Float16* p = s + (row0 + (lane & 15)) * stride + col0 + ((lane >> 4) << 3);
  v8h lo = *(const v8h*)(p);
  v8h hi = *(const v8h*)(p + 16);
  return __builtin_shufflevector(lo, hi, 0, 1, 2, 3, 4, 5, 6, 7,
                                 8, 9, 10, 11, 12, 13, 14, 15);
}

// B fragments are stored PRE-SWIZZLED (fragment-major): for fragment f,
// lane L's 16 halves live contiguously at s[(f*32 + L)*16 .. +15].
__device__ inline v16h frag_B_swz(const _Float16* s, int fragIdx, int lane) {
  return *(const v16h*)(s + (((fragIdx << 5) + lane) << 4));
}

// Swizzle source element for fragment-major B storage.
// Register layout: lane L holds column N = nt*16 + L%16,
//   half j = W[ks*32 + 8*(L/16) + (j&7) + 16*(j>>3)][N]   (zero beyond realK rows).
__device__ inline _Float16 swz_elem(const float* __restrict__ W, int i, int realK) {
  const int frag = i >> 9;          // fragment index (ks*8 + nt)
  const int w    = i & 511;
  const int lane = w >> 4;
  const int j    = w & 15;
  const int ks   = frag >> 3;
  const int nt   = frag & 7;
  const int row  = ks * 32 + ((lane >> 4) << 3) + (j & 7) + ((j >> 3) << 4);
  const int col  = nt * 16 + (lane & 15);
  return (row < realK) ? (_Float16)W[row * OUT_ + col] : (_Float16)0.f;
}

// ---------- Kernel 0: f32 -> f16 weight conversion + fragment swizzle ----------
__global__ __launch_bounds__(256) void convert_weights_kernel(
    const float* __restrict__ W1, const float* __restrict__ W2,
    const float* __restrict__ Wn, _Float16* __restrict__ W1s,
    _Float16* __restrict__ W2s, _Float16* __restrict__ Wns) {
  int i = blockIdx.x * 256 + threadIdx.x;            // covers 0 .. 288*128-1
  if (i < K1P * OUT_) W1s[i] = swz_elem(W1, i, D_ + 7);        // 135 real rows
  if (i < K2P * OUT_) W2s[i] = swz_elem(W2, i, K2P);           // 128 real rows
  if (i < KNP * OUT_) Wns[i] = swz_elem(Wn, i, 2 * D_ + CD_);  // 261 real rows
}

// ---------- Kernel 1: edge MLP + masked aggregation (1 workgroup per node) ----------
__global__ __launch_bounds__(256) void edge_mlp_kernel(
    const float* __restrict__ h, const float* __restrict__ pos,
    const float* __restrict__ vel, const float* __restrict__ acc_,
    const float* __restrict__ crowd, const int* __restrict__ mask,
    const int* __restrict__ idex, const float* __restrict__ b1,
    const float* __restrict__ b2, const _Float16* __restrict__ W1s,
    const _Float16* __restrict__ W2s, float* __restrict__ agg) {
  __shared__ __align__(32) _Float16 sA[K_ * K1P];   // 64x160 edge input; reused 64x128 hidden
  __shared__ __align__(32) _Float16 sW[K1P * OUT_]; // swizzled W1 (40 frags); reused W2 (32 frags)
  __shared__ float sMask[K_];
  __shared__ float sAgg[OUT_];
  __shared__ float sB1[OUT_], sB2[OUT_];
  __shared__ float sInv;

  const int tid  = threadIdx.x;
  const int lane = tid & 31;
  const int wave = tid >> 5;
  const int node = blockIdx.x;       // b*N + n
  const int b    = node / N_;

  // ---- Phase 1: stage swizzled W1 (async) + biases, zero agg ----
  stage_g2l(W1s, sW, K1P * OUT_ * 2, tid);
  if (tid < OUT_) { sB1[tid] = b1[tid]; sB2[tid] = b2[tid]; sAgg[tid] = 0.f; }

  // ---- Phase 2: build 64x160 edge-input tile (4 threads per edge) ----
  {
    const int e   = tid >> 2;
    const int sub = tid & 3;
    const long eb = (long)node * K_ + e;
    const int  mk = mask[eb];
    const int  m  = idex[eb] * mk;
    const long nb = (long)b * N_ + m;

    __builtin_prefetch(h + nb * D_, 0, 3);

    const float4* hs = (const float4*)(h + nb * D_ + sub * 32);
    _Float16* da = sA + e * K1P + sub * 32;
#pragma unroll
    for (int j = 0; j < 8; ++j) {
      float4 v = hs[j];
      da[4 * j + 0] = (_Float16)v.x; da[4 * j + 1] = (_Float16)v.y;
      da[4 * j + 2] = (_Float16)v.z; da[4 * j + 3] = (_Float16)v.w;
    }
    if (sub == 0) {
      const long pb = (long)node;
      float px = pos[pb * 2 + 0], py = pos[pb * 2 + 1];
      float vx = vel[pb * 2 + 0], vy = vel[pb * 2 + 1];
      float ax = acc_[pb * 2 + 0], ay = acc_[pb * 2 + 1];
      float ivn = 1.f / fmaxf(sqrtf(vx * vx + vy * vy), 1e-12f);
      float vix = vx * ivn, viy = vy * ivn;
      float c0 = crowd[pb * CD_ + 0], c1 = crowd[pb * CD_ + 1];
      float c2 = crowd[pb * CD_ + 2], c3 = crowd[pb * CD_ + 3];
      float pn = sqrtf(vx * vx + vy * vy + ax * ax + ay * ay);
      float cn = sqrtf(c0 * c0 + c1 * c1 + c2 * c2 + c3 * c3);
      float cos_sim = (vx * c0 + vy * c1 + ax * c2 + ay * c3) / (pn * cn + 1e-6f);
      float crowd_sim = (cos_sim + 1.f) * 0.5f;

      float npx = pos[nb * 2 + 0], npy = pos[nb * 2 + 1];
      float nvx = vel[nb * 2 + 0], nvy = vel[nb * 2 + 1];
      float rx = npx - px, ry = npy - py;
      float rn = sqrtf(rx * rx + ry * ry);
      float dist = rn + 1e-6f;
      float irn = 1.f / fmaxf(rn, 1e-12f);
      float rdx = rx * irn, rdy = ry * irn;
      float invn = 1.f / fmaxf(sqrtf(nvx * nvx + nvy * nvy), 1e-12f);
      float vjx = nvx * invn, vjy = nvy * invn;
      float fs1 = (rdx * vjx + rdy * vjy + 1.f) * 0.5f;
      float fs2 = (vix * vjx + viy * vjy + 1.f) * 0.5f;
      float dvx = vx - nvx, dvy = vy - nvy;
      float rsp = sqrtf(dvx * dvx + dvy * dvy);

      _Float16* f = sA + e * K1P + D_;
      f[0] = (_Float16)rx;  f[1] = (_Float16)ry;  f[2] = (_Float16)dist;
      f[3] = (_Float16)fs1; f[4] = (_Float16)fs2; f[5] = (_Float16)crowd_sim;
      f[6] = (_Float16)rsp;
      sMask[e] = (float)mk;
    } else if (sub == 1) {
      _Float16* f = sA + e * K1P + D_ + 7;
      for (int j = 0; j < K1P - D_ - 7; ++j) f[j] = (_Float16)0.f;
    }
  }
  stage_wait();
  __syncthreads();
  if (tid == 0) {
    float s = 0.f;
    for (int e = 0; e < K_; ++e) s += sMask[e];
    sInv = 1.f / (s + 1e-6f);
  }

  // 32 C-tiles (4 M x 8 N), 4 per wave (same M-tile -> A-frag reuse)
  const int idx0 = wave * 4;
  const int mi   = idx0 >> 3;
  const int ni0  = idx0 & 7;
  const v8f vzero = {0.f, 0.f, 0.f, 0.f, 0.f, 0.f, 0.f, 0.f};

  // ---- GEMM1: (64x160) @ (160x128) ----
  v8f acc1[4];
#pragma unroll
  for (int t = 0; t < 4; ++t) acc1[t] = vzero;
#pragma unroll
  for (int ks = 0; ks < K1P / 32; ++ks) {
    v16h a = frag_A(sA, K1P, mi * 16, ks * 32, lane);
#pragma unroll
    for (int t = 0; t < 4; ++t) {
      v16h bf = frag_B_swz(sW, ks * 8 + ni0 + t, lane);
      acc1[t] = __builtin_amdgcn_wmma_f32_16x16x32_f16(false, a, false, bf,
                                                       (short)0, acc1[t], false, false);
    }
  }
  __syncthreads();

  // bias + relu -> hidden tile (reuse sA region, stride 128) ; restage W2 (async)
  {
    const int rowb = mi * 16 + ((lane >> 4) << 3);
    const int colb = lane & 15;
#pragma unroll
    for (int t = 0; t < 4; ++t) {
      const int col = (ni0 + t) * 16 + colb;
      const float bb = sB1[col];
#pragma unroll
      for (int v = 0; v < 8; ++v)
        sA[(rowb + v) * K2P + col] = (_Float16)fmaxf(acc1[t][v] + bb, 0.f);
    }
  }
  stage_g2l(W2s, sW, K2P * OUT_ * 2, tid);
  stage_wait();
  __syncthreads();

  // ---- GEMM2: (64x128) @ (128x128) ----
  v8f acc2[4];
#pragma unroll
  for (int t = 0; t < 4; ++t) acc2[t] = vzero;
#pragma unroll
  for (int ks = 0; ks < K2P / 32; ++ks) {
    v16h a = frag_A(sA, K2P, mi * 16, ks * 32, lane);
#pragma unroll
    for (int t = 0; t < 4; ++t) {
      v16h bf = frag_B_swz(sW, ks * 8 + ni0 + t, lane);
      acc2[t] = __builtin_amdgcn_wmma_f32_16x16x32_f16(false, a, false, bf,
                                                       (short)0, acc2[t], false, false);
    }
  }

  // mask-weighted column reduction over the 64 edge rows
  {
    const int rowb = mi * 16 + ((lane >> 4) << 3);
    const int colb = lane & 15;
#pragma unroll
    for (int t = 0; t < 4; ++t) {
      const int col = (ni0 + t) * 16 + colb;
      const float bb = sB2[col];
      float s = 0.f;
#pragma unroll
      for (int v = 0; v < 8; ++v) s += (acc2[t][v] + bb) * sMask[rowb + v];
      atomicAdd(&sAgg[col], s);
    }
  }
  __syncthreads();
  if (tid < OUT_) agg[(long)node * OUT_ + tid] = sAgg[tid] * sInv;
}

// ---------- Kernel 2: node MLP (64 nodes per workgroup) ----------
__global__ __launch_bounds__(256) void node_mlp_kernel(
    const float* __restrict__ h, const float* __restrict__ crowd,
    const float* __restrict__ ln_g, const float* __restrict__ ln_b,
    const float* __restrict__ bn, const _Float16* __restrict__ Wns,
    const float* __restrict__ agg, float* __restrict__ out) {
  __shared__ __align__(32) _Float16 sA[64 * KNP];   // 64 x 288 node input
  __shared__ __align__(32) _Float16 sW[96 * OUT_];  // swizzled Wn chunk (24 frags)
  __shared__ float sBn[OUT_];

  const int tid  = threadIdx.x;
  const int lane = tid & 31;
  const int wave = tid >> 5;
  const long row0 = (long)blockIdx.x * 64;

  if (tid < OUT_) sBn[tid] = bn[tid];

  // build node-input tile: [h(128) | agg(128) | layernorm(crowd)(5) | zero-pad]
  {
    const int lr  = tid >> 2;
    const int sub = tid & 3;
    const long row = row0 + lr;
    _Float16* da = sA + lr * KNP + sub * 64;
    const float* src = (sub < 2) ? (h + row * D_ + sub * 64)
                                 : (agg + row * OUT_ + (sub - 2) * 64);
    const float4* s4 = (const float4*)src;
#pragma unroll
    for (int j = 0; j < 16; ++j) {
      float4 v = s4[j];
      da[4 * j + 0] = (_Float16)v.x; da[4 * j + 1] = (_Float16)v.y;
      da[4 * j + 2] = (_Float16)v.z; da[4 * j + 3] = (_Float16)v.w;
    }
    if (sub == 0) {
      float c[CD_]; float mu = 0.f;
#pragma unroll
      for (int j = 0; j < CD_; ++j) { c[j] = crowd[row * CD_ + j]; mu += c[j]; }
      mu *= (1.f / CD_);
      float var = 0.f;
#pragma unroll
      for (int j = 0; j < CD_; ++j) { float d = c[j] - mu; var += d * d; }
      var *= (1.f / CD_);
      float inv = 1.f / sqrtf(var + 1e-5f);
      _Float16* f = sA + lr * KNP + 2 * D_;
#pragma unroll
      for (int j = 0; j < CD_; ++j)
        f[j] = (_Float16)(((c[j] - mu) * inv) * ln_g[j] + ln_b[j]);
    } else if (sub == 1) {
      _Float16* f = sA + lr * KNP + 2 * D_ + CD_;
      for (int j = 0; j < KNP - 2 * D_ - CD_; ++j) f[j] = (_Float16)0.f;
    }
  }

  const int idx0 = wave * 4;
  const int mi   = idx0 >> 3;
  const int ni0  = idx0 & 7;
  const v8f vzero = {0.f, 0.f, 0.f, 0.f, 0.f, 0.f, 0.f, 0.f};
  v8f accn[4];
#pragma unroll
  for (int t = 0; t < 4; ++t) accn[t] = vzero;

  // ---- GEMM: (64x288) @ (288x128), swizzled Wn streamed through LDS, 3 ksteps/chunk ----
  for (int chunk = 0; chunk < 3; ++chunk) {
    __syncthreads();
    stage_g2l(Wns + chunk * 96 * OUT_, sW, 96 * OUT_ * 2, tid);
    stage_wait();
    __syncthreads();
#pragma unroll
    for (int ks = 0; ks < 3; ++ks) {
      v16h a = frag_A(sA, KNP, mi * 16, (chunk * 3 + ks) * 32, lane);
#pragma unroll
      for (int t = 0; t < 4; ++t) {
        v16h bf = frag_B_swz(sW, ks * 8 + ni0 + t, lane);
        accn[t] = __builtin_amdgcn_wmma_f32_16x16x32_f16(false, a, false, bf,
                                                         (short)0, accn[t], false, false);
      }
    }
  }

  // bias + relu -> output (f32)
  {
    const int rowb = mi * 16 + ((lane >> 4) << 3);
    const int colb = lane & 15;
#pragma unroll
    for (int t = 0; t < 4; ++t) {
      const int col = (ni0 + t) * 16 + colb;
      const float bb = sBn[col];
#pragma unroll
      for (int v = 0; v < 8; ++v)
        out[(row0 + rowb + v) * OUT_ + col] = fmaxf(accn[t][v] + bb, 0.f);
    }
  }
}

extern "C" void kernel_launch(void* const* d_in, const int* in_sizes, int n_in,
                              void* d_out, int out_size, void* d_ws, size_t ws_size,
                              hipStream_t stream) {
  const float* h     = (const float*)d_in[0];
  const float* pos   = (const float*)d_in[1];
  const float* vel   = (const float*)d_in[2];
  const float* acc   = (const float*)d_in[3];
  const float* crowd = (const float*)d_in[4];
  const int*   mask  = (const int*)d_in[5];
  const int*   idex  = (const int*)d_in[6];
  const float* W1    = (const float*)d_in[8];
  const float* b1    = (const float*)d_in[9];
  const float* W2    = (const float*)d_in[10];
  const float* b2    = (const float*)d_in[11];
  const float* Wn    = (const float*)d_in[12];
  const float* bn    = (const float*)d_in[13];
  const float* ln_g  = (const float*)d_in[14];
  const float* ln_b  = (const float*)d_in[15];

  char* ws = (char*)d_ws;
  _Float16* W1s = (_Float16*)(ws);                           // 160*128*2 = 40960 B
  _Float16* W2s = (_Float16*)(ws + 40960);                   // 128*128*2 = 32768 B
  _Float16* Wns = (_Float16*)(ws + 40960 + 32768);           // 288*128*2 = 73728 B
  float*    agg = (float*)(ws + 40960 + 32768 + 73728);      // 4096*128*4 = 2 MB

  convert_weights_kernel<<<(KNP * OUT_) / 256, 256, 0, stream>>>(W1, W2, Wn, W1s, W2s, Wns);
  edge_mlp_kernel<<<B_ * N_, 256, 0, stream>>>(h, pos, vel, acc, crowd, mask, idex,
                                               b1, b2, W1s, W2s, agg);
  node_mlp_kernel<<<(B_ * N_) / 64, 256, 0, stream>>>(h, crowd, ln_g, ln_b, bn, Wns,
                                                      agg, (float*)d_out);
}
